// EfficientEquilibriumLayer_14817637171726
// MI455X (gfx1250) — compile-verified, matrix operand order checked by reading, not run on hardware
//
#include <hip/hip_runtime.h>
#include <cstdint>
#include <cstddef>

typedef __bf16 bf16;
typedef __attribute__((ext_vector_type(16))) __bf16 bf16x16;
typedef __attribute__((ext_vector_type(8)))  float  f32x8;

// ---------------------------------------------------------------------------
// WMMA helpers (CDNA5: D = A(16x32 bf16) x B(32x16 bf16) + C(16x16 f32))
// ---------------------------------------------------------------------------
__device__ __forceinline__ f32x8 wmma_bf16(bf16x16 a, bf16x16 b, f32x8 c) {
  return __builtin_amdgcn_wmma_f32_16x16x32_bf16(false, a, false, b, (short)0, c,
                                                 false, false);
}

// Fragment loader for A (MxK) and B (NxK) operands. Per ISA 7.12.2 (16-bit
// 16x32): lanes 0-15 hold row (lane&15) with K halves {0..7, 16..23};
// lanes 16-31 hold {8..15, 24..31}. rs = row stride, ks = K stride.
__device__ __forceinline__ bf16x16 frag_ld(const bf16* p, int rs, int ks) {
  const int lane = threadIdx.x & 31;
  const int r = lane & 15;
  const int kb = (lane >> 4) << 3;  // 0 or 8
  bf16x16 f;
#pragma unroll
  for (int i = 0; i < 8; ++i) f[i] = p[r * rs + (kb + i) * ks];
#pragma unroll
  for (int i = 0; i < 8; ++i) f[8 + i] = p[r * rs + (kb + 16 + i) * ks];
  return f;
}

// ---------------------------------------------------------------------------
// Async memory->LDS copy (CDNA5 GLOBAL_LOAD_ASYNC_TO_LDS_B128, ASYNCcnt).
// Low 32 bits of a generic pointer to __shared__ are the LDS byte address
// (ISA 10.2: LDS generic address = {aperture, offset[31:0]}).
// ---------------------------------------------------------------------------
__device__ __forceinline__ void async_copy_b128(void* lds, const void* g) {
  asm volatile("global_load_async_to_lds_b128 %0, %1, off"
               :: "v"((uint32_t)(uintptr_t)lds), "v"((uint64_t)(uintptr_t)g)
               : "memory");
}
__device__ __forceinline__ void wait_async0() {
  asm volatile("s_wait_asynccnt 0" ::: "memory");
}

// ---------------------------------------------------------------------------
// fp32 -> bf16 weight conversion
// ---------------------------------------------------------------------------
__global__ void cvt_kernel(const float* __restrict__ s, bf16* __restrict__ d, int n) {
  int i = blockIdx.x * blockDim.x + threadIdx.x;
  if (i < n) d[i] = (bf16)s[i];
}

// ---------------------------------------------------------------------------
// LayerNorm over D=1024, one row per block (256 threads), bf16 output
// ---------------------------------------------------------------------------
__global__ __launch_bounds__(256) void ln_kernel(const float* __restrict__ x,
                                                 const float* __restrict__ w,
                                                 const float* __restrict__ b,
                                                 bf16* __restrict__ out) {
  const int row = blockIdx.x;
  const int tid = threadIdx.x;
  const float* xr = x + (size_t)row * 1024;
  float v[4];
  float s = 0.f, sq = 0.f;
#pragma unroll
  for (int i = 0; i < 4; ++i) {
    v[i] = xr[tid + 256 * i];
    s += v[i];
    sq += v[i] * v[i];
  }
  __shared__ float rs[256], rq[256];
  rs[tid] = s; rq[tid] = sq;
  __syncthreads();
  for (int o = 128; o > 0; o >>= 1) {
    if (tid < o) { rs[tid] += rs[tid + o]; rq[tid] += rq[tid + o]; }
    __syncthreads();
  }
  const float mu = rs[0] * (1.f / 1024.f);
  const float var = rq[0] * (1.f / 1024.f) - mu * mu;
  const float inv = rsqrtf(var + 1e-5f);
#pragma unroll
  for (int i = 0; i < 4; ++i) {
    int c = tid + 256 * i;
    out[(size_t)row * 1024 + c] = (bf16)((v[i] - mu) * inv * w[c] + b[c]);
  }
}

// ---------------------------------------------------------------------------
// WMMA GEMM: C[M,N] = act(A[M,K] @ W[N,K]^T + bias) (+ resid), bf16/f32 out.
// 256 threads (8 wave32), tile 128x64, K step 32. LDS double-buffered via
// async-to-LDS copies: issue next slab, WMMA current slab, wait, barrier.
// ---------------------------------------------------------------------------
enum { ACT_NONE = 0, ACT_ELU1 = 1, ACT_GELU = 2 };

template <int ACT>
__global__ __launch_bounds__(256) void gemm_kernel(
    const bf16* __restrict__ A, const bf16* __restrict__ W,
    const float* __restrict__ bias, float* __restrict__ outF,
    bf16* __restrict__ outB, const float* __restrict__ resid,
    int M, int N, int K) {
  constexpr int LDA = 40;  // padded ld (bank-conflict-free b128 LDS reads)
  __shared__ bf16 sA[2][128 * LDA];
  __shared__ bf16 sW[2][64 * LDA];
  const int tid = threadIdx.x;
  const int wave = tid >> 5;
  const int wm = wave >> 1;  // 0..3 (M)
  const int wn = wave & 1;   // 0..1 (N)
  const int m0 = blockIdx.y * 128;
  const int n0 = blockIdx.x * 64;

  // fixed per-thread staging coordinates (16B chunks)
  const int ar0 = (tid * 2) >> 2, ac0 = ((tid * 2) & 3) * 8;
  const int ar1 = (tid * 2 + 1) >> 2, ac1 = ((tid * 2 + 1) & 3) * 8;
  const int wr = tid >> 2, wc = (tid & 3) * 8;

  auto issue = [&](int buf, int k0) {
    async_copy_b128(&sA[buf][ar0 * LDA + ac0], &A[(size_t)(m0 + ar0) * K + k0 + ac0]);
    async_copy_b128(&sA[buf][ar1 * LDA + ac1], &A[(size_t)(m0 + ar1) * K + k0 + ac1]);
    async_copy_b128(&sW[buf][wr * LDA + wc],   &W[(size_t)(n0 + wr) * K + k0 + wc]);
  };

  issue(0, 0);
  wait_async0();
  __syncthreads();

  f32x8 acc[2][2] = {};
  const int nk = K / 32;
  for (int i = 0; i < nk; ++i) {
    const int cur = i & 1;
    if (i + 1 < nk) issue(cur ^ 1, (i + 1) * 32);  // prefetch next slab

    bf16x16 af[2], wf[2];
#pragma unroll
    for (int a = 0; a < 2; ++a)
      af[a] = frag_ld(&sA[cur][(wm * 32 + a * 16) * LDA], LDA, 1);
#pragma unroll
    for (int j = 0; j < 2; ++j)
      wf[j] = frag_ld(&sW[cur][(wn * 32 + j * 16) * LDA], LDA, 1);
#pragma unroll
    for (int a = 0; a < 2; ++a)
#pragma unroll
      for (int j = 0; j < 2; ++j) acc[a][j] = wmma_bf16(af[a], wf[j], acc[a][j]);

    wait_async0();   // own prefetch complete
    __syncthreads(); // everyone's prefetch complete + compute done
  }

  // Epilogue. C layout: lane holds col (lane&15); rows = vgpr + 8*(lane>=16).
  const int lane = tid & 31;
  const int cN = lane & 15;
  const int r8 = (lane >> 4) << 3;
#pragma unroll
  for (int a = 0; a < 2; ++a)
#pragma unroll
    for (int j = 0; j < 2; ++j) {
      const int col = n0 + wn * 32 + j * 16 + cN;
      const float bv = bias ? bias[col] : 0.f;
#pragma unroll
      for (int v = 0; v < 8; ++v) {
        const int row = m0 + wm * 32 + a * 16 + r8 + v;
        float xv = acc[a][j][v] + bv;
        if (ACT == ACT_ELU1) xv = xv > 0.f ? xv + 1.f : __expf(xv);
        else if (ACT == ACT_GELU) xv = 0.5f * xv * (1.f + erff(xv * 0.70710678118f));
        const size_t idx = (size_t)row * N + col;
        if (resid) xv += resid[idx];
        if (outF) outF[idx] = xv;
        if (outB) outB[idx] = (bf16)xv;
      }
    }
}

// ---------------------------------------------------------------------------
// Chunked causal linear attention. One block per (batch, 64-wide D-tile).
// Per 64-token chunk: S = mask(Q K^T); den = rowsum(S) + Q.Ks;
// num = S V + Q KV; out = num/den; KV += K^T V; Ks += colsum(K).
// All four 64x64x64 GEMMs done with bf16 WMMA; state in LDS (f32 + bf16).
// ---------------------------------------------------------------------------
__global__ __launch_bounds__(256) void attn_kernel(const bf16* __restrict__ Q,
                                                   const bf16* __restrict__ K,
                                                   const bf16* __restrict__ V,
                                                   bf16* __restrict__ O) {
  constexpr int LD = 72, LDKV = 65;
  __shared__ bf16 sQ[64 * LD], sK[64 * LD], sV[64 * LD], sS[64 * LD], sKVb[64 * LD];
  __shared__ float sKV[64 * LDKV];
  __shared__ float sKs[64], sDen[64];
  const int tid = threadIdx.x;
  const int wave = tid >> 5, lane = tid & 31;
  const int b = blockIdx.y;
  const int d0 = blockIdx.x * 64;
  const int cN = lane & 15, r8 = (lane >> 4) << 3;

  for (int i = tid; i < 64 * LDKV; i += 256) sKV[i] = 0.f;
  for (int i = tid; i < 64 * LD; i += 256) sKVb[i] = (bf16)0.f;
  if (tid < 64) sKs[tid] = 0.f;
  __syncthreads();

  for (int c = 0; c < 64; ++c) {
    const size_t tok0 = (size_t)b * 4096 + (size_t)c * 64;
    // stage Q,K (ld 64) and V slice (ld 1024) via async-to-LDS copies
#pragma unroll
    for (int i = 0; i < 2; ++i) {
      int ch = tid * 2 + i;
      int r = ch >> 3, cc = (ch & 7) * 8;
      async_copy_b128(&sQ[r * LD + cc], &Q[(tok0 + r) * 64 + cc]);
      async_copy_b128(&sK[r * LD + cc], &K[(tok0 + r) * 64 + cc]);
      async_copy_b128(&sV[r * LD + cc], &V[(tok0 + r) * 1024 + d0 + cc]);
    }
    wait_async0();
    __syncthreads();

    // S = Q @ K^T (16 tiles over 8 waves), causal mask, bf16 into sS
#pragma unroll
    for (int t = 0; t < 2; ++t) {
      const int tl = wave * 2 + t, tm = tl >> 2, tn = tl & 3;
      f32x8 acc = {};
#pragma unroll
      for (int kk = 0; kk < 2; ++kk)
        acc = wmma_bf16(frag_ld(&sQ[tm * 16 * LD + kk * 32], LD, 1),
                        frag_ld(&sK[tn * 16 * LD + kk * 32], LD, 1), acc);
#pragma unroll
      for (int v = 0; v < 8; ++v) {
        const int row = tm * 16 + r8 + v, col = tn * 16 + cN;
        sS[row * LD + col] = (bf16)(col <= row ? acc[v] : 0.f);
      }
    }
    __syncthreads();

    // den_i = eps + q_i . Ks_state + rowsum(S_i)
    if (tid < 64) {
      float s = 1e-6f;
#pragma unroll 8
      for (int f = 0; f < 64; ++f) s += (float)sQ[tid * LD + f] * sKs[f];
#pragma unroll 8
      for (int j = 0; j < 64; ++j) s += (float)sS[tid * LD + j];
      sDen[tid] = s;
    }
    __syncthreads();

    // num = S @ V + Q @ KV ; out = num / den
#pragma unroll
    for (int t = 0; t < 2; ++t) {
      const int tl = wave * 2 + t, tm = tl >> 2, tn = tl & 3;
      f32x8 acc = {};
#pragma unroll
      for (int kk = 0; kk < 2; ++kk)  // intra-chunk: B = V^T access
        acc = wmma_bf16(frag_ld(&sS[tm * 16 * LD + kk * 32], LD, 1),
                        frag_ld(&sV[tn * 16 + kk * 32 * LD], 1, LD), acc);
#pragma unroll
      for (int kk = 0; kk < 2; ++kk)  // inter-chunk: B = KV^T access
        acc = wmma_bf16(frag_ld(&sQ[tm * 16 * LD + kk * 32], LD, 1),
                        frag_ld(&sKVb[tn * 16 + kk * 32 * LD], 1, LD), acc);
#pragma unroll
      for (int v = 0; v < 8; ++v) {
        const int row = tm * 16 + r8 + v, col = tn * 16 + cN;
        O[(tok0 + row) * 1024 + d0 + col] = (bf16)(acc[v] / sDen[row]);
      }
    }
    __syncthreads();

    // KV_state += K^T @ V ; Ks_state += colsum(K); refresh bf16 mirror
#pragma unroll
    for (int t = 0; t < 2; ++t) {
      const int tl = wave * 2 + t, tm = tl >> 2, tn = tl & 3;
      f32x8 acc = {};
#pragma unroll
      for (int kk = 0; kk < 2; ++kk)
        acc = wmma_bf16(frag_ld(&sK[tm * 16 + kk * 32 * LD], 1, LD),
                        frag_ld(&sV[tn * 16 + kk * 32 * LD], 1, LD), acc);
#pragma unroll
      for (int v = 0; v < 8; ++v) {
        const int row = tm * 16 + r8 + v, col = tn * 16 + cN;
        const float nv = sKV[row * LDKV + col] + acc[v];
        sKV[row * LDKV + col] = nv;
        sKVb[row * LD + col] = (bf16)nv;
      }
    }
    if (tid < 64) {
      float s = 0.f;
#pragma unroll 8
      for (int t = 0; t < 64; ++t) s += (float)sK[t * LD + tid];
      sKs[tid] += s;
    }
    __syncthreads();
  }
}

// ---------------------------------------------------------------------------
// Host-side orchestration
// ---------------------------------------------------------------------------
extern "C" void kernel_launch(void* const* d_in, const int* in_sizes, int n_in,
                              void* d_out, int out_size, void* d_ws, size_t ws_size,
                              hipStream_t stream) {
  const float* x    = (const float*)d_in[0];
  const float* ln1w = (const float*)d_in[1];
  const float* ln1b = (const float*)d_in[2];
  const float* qw   = (const float*)d_in[3];
  const float* qb   = (const float*)d_in[4];
  const float* kw   = (const float*)d_in[5];
  const float* kb   = (const float*)d_in[6];
  const float* vw   = (const float*)d_in[7];
  const float* vb   = (const float*)d_in[8];
  const float* ow   = (const float*)d_in[9];
  const float* ob   = (const float*)d_in[10];
  const float* ln2w = (const float*)d_in[11];
  const float* ln2b = (const float*)d_in[12];
  const float* f1w  = (const float*)d_in[13];
  const float* f1b  = (const float*)d_in[14];
  const float* f2w  = (const float*)d_in[15];
  const float* f2b  = (const float*)d_in[16];

  const int M = 4 * 4096, D = 1024, F = 64, H = 4096;

  char* ws = (char*)d_ws;
  size_t off = 0;
  auto alloc = [&](size_t bytes) -> void* {
    void* p = ws + off;
    off = (off + bytes + 255) & ~(size_t)255;
    return p;
  };
  bf16*  XN  = (bf16*)alloc((size_t)M * D * 2);  // ln1 out; reused as attn out
  bf16*  Qb_ = (bf16*)alloc((size_t)M * F * 2);
  bf16*  Kb_ = (bf16*)alloc((size_t)M * F * 2);
  bf16*  Vb_ = (bf16*)alloc((size_t)M * D * 2);  // V; reused as ln2 out
  float* X1  = (float*)alloc((size_t)M * D * 4);
  bf16*  G   = (bf16*)alloc((size_t)M * H * 2);
  bf16*  wQ  = (bf16*)alloc((size_t)F * D * 2);
  bf16*  wK  = (bf16*)alloc((size_t)F * D * 2);
  bf16*  wV  = (bf16*)alloc((size_t)D * D * 2);
  bf16*  wO  = (bf16*)alloc((size_t)D * D * 2);
  bf16*  wF1 = (bf16*)alloc((size_t)H * D * 2);
  bf16*  wF2 = (bf16*)alloc((size_t)D * H * 2);

  auto cvt = [&](const float* s, bf16* d, int n) {
    cvt_kernel<<<(n + 255) / 256, 256, 0, stream>>>(s, d, n);
  };
  cvt(qw, wQ, F * D);
  cvt(kw, wK, F * D);
  cvt(vw, wV, D * D);
  cvt(ow, wO, D * D);
  cvt(f1w, wF1, H * D);
  cvt(f2w, wF2, D * H);

  // x_norm = LN1(x)
  ln_kernel<<<M, 256, 0, stream>>>(x, ln1w, ln1b, XN);

  // Q/K = elu(lin)+1 ; V = lin
  gemm_kernel<ACT_ELU1><<<dim3(F / 64, M / 128), 256, 0, stream>>>(
      XN, wQ, qb, nullptr, Qb_, nullptr, M, F, D);
  gemm_kernel<ACT_ELU1><<<dim3(F / 64, M / 128), 256, 0, stream>>>(
      XN, wK, kb, nullptr, Kb_, nullptr, M, F, D);
  gemm_kernel<ACT_NONE><<<dim3(D / 64, M / 128), 256, 0, stream>>>(
      XN, wV, vb, nullptr, Vb_, nullptr, M, D, D);

  // chunked causal linear attention (writes over XN)
  attn_kernel<<<dim3(D / 64, 4), 256, 0, stream>>>(Qb_, Kb_, Vb_, XN);

  // x1 = x + O(attn)
  gemm_kernel<ACT_NONE><<<dim3(D / 64, M / 128), 256, 0, stream>>>(
      XN, wO, ob, X1, nullptr, x, M, D, D);

  // h = LN2(x1)  (into Vb_ storage)
  ln_kernel<<<M, 256, 0, stream>>>(X1, ln2w, ln2b, Vb_);

  // g = gelu(F1(h))
  gemm_kernel<ACT_GELU><<<dim3(H / 64, M / 128), 256, 0, stream>>>(
      Vb_, wF1, f1b, nullptr, G, nullptr, M, H, D);

  // out = x1 + F2(g)
  gemm_kernel<ACT_NONE><<<dim3(D / 64, M / 128), 256, 0, stream>>>(
      G, wF2, f2b, (float*)d_out, nullptr, X1, M, D, H);
}
